// MultiHeadSelfAttention_9904194584967
// MI455X (gfx1250) — matrix operand order repaired and checked by reading, not compiled, not run on hardware
//
#include <hip/hip_runtime.h>

// ---------------------------------------------------------------- types ----
typedef __bf16 bf16_t;
typedef __attribute__((ext_vector_type(16))) __bf16 v16bf;
typedef __attribute__((ext_vector_type(8)))  __bf16 v8bf;
typedef __attribute__((ext_vector_type(4)))  __bf16 v4bf;
typedef __attribute__((ext_vector_type(8)))  float  v8f;
typedef __attribute__((ext_vector_type(4)))  float  v4f;

// ---------------------------------------------------------------- shapes ---
#define Bb 4
#define Nn 2048
#define Cc 768
#define Hh 12
#define Dd 64
#define QKV_ELEMS ((size_t)Bb * Hh * Nn * Dd)   // one of q/k/v

// ---------------------------------------------------------------- tiling ---
#define BM 128         // block tile rows
#define BN 64          // block tile cols
#define BK 32          // K step (all K used are even multiples of BK)
#define AS 40          // LDS row stride (bf16): 80B -> 16B aligned, conflict-free frags
#define BS 40
#define BUFE (BM * AS + BN * BS)   // elems per LDS buffer (double buffered)
#define NTHREADS 256   // 8 waves: 4 (row) x 2 (col); wave tile 32x32 = 4 WMMAs

// ---------------------------------------------------------------------------
// Double-buffered WMMA GEMM tile engine.
//   A : [M,K] f32 row-major (converted to bf16 on the fill path)
//   Bm: [Ncols,K] bf16 row-major (pre-transposed "[n][k]-major")
// Per k-step: B tile -> global_load_async_to_lds_b128 (ASYNCcnt);
// A tile -> 4x global_load_b128 staged in regs, v_cvt_pk_bf16_f32, ds_store_b64
// AFTER the WMMAs so global latency hides behind matrix math.
// ---------------------------------------------------------------------------
struct GemmCtx {
    const float* Abase;      // A + (tileM + tid/8)*lda + (tid%8)*4
    const bf16_t* Bbase;     // Bm + (tileN + tid/4)*ldb + (tid%4)*8
    size_t lda;              // row stride of A (elems)
    size_t ldb;              // row stride of B (elems)
    int afill_off;           // LDS elem offset for this thread's A chunk (row 0)
    int bfill_off;           // LDS elem offset for this thread's B chunk
};

static __device__ inline void issue_fill(const GemmCtx& c, int kb, bf16_t* buf,
                                         v4f va[4]) {
    // B tile: 64n x 32k bf16 = 4KB, one 16B async chunk per lane
    {
        const bf16_t* g = c.Bbase + kb;
        unsigned loff = (unsigned)(unsigned long long)(&buf[BM * AS + c.bfill_off]);
        asm volatile("global_load_async_to_lds_b128 %0, %1, off"
                     :: "v"(loff), "v"(g) : "memory");
    }
    // A tile: stage 4 float4 loads (kept in flight together)
    #pragma unroll
    for (int i = 0; i < 4; ++i)
        va[i] = *(const v4f*)(c.Abase + kb + (size_t)i * 32 * c.lda);
}

static __device__ inline void store_fill(const GemmCtx& c, bf16_t* buf, v4f va[4]) {
    #pragma unroll
    for (int i = 0; i < 4; ++i)
        *(v4bf*)&buf[c.afill_off + i * 32 * AS] = __builtin_convertvector(va[i], v4bf);
}

static __device__ inline void compute_step(const bf16_t* buf, int wr, int wc,
                                           int ll, int lh, v8f acc[2][2]) {
    const bf16_t* Alds = buf;
    const bf16_t* Blds = buf + BM * AS;
    v16bf a[2], b[2];
    #pragma unroll
    for (int i = 0; i < 2; ++i) {
        const int arow = wr * 32 + i * 16 + ll;
        v8bf lo = *reinterpret_cast<const v8bf*>(&Alds[arow * AS + 8 * lh]);
        v8bf hi = *reinterpret_cast<const v8bf*>(&Alds[arow * AS + 16 + 8 * lh]);
        a[i] = __builtin_shufflevector(lo, hi, 0,1,2,3,4,5,6,7,8,9,10,11,12,13,14,15);
    }
    #pragma unroll
    for (int j = 0; j < 2; ++j) {
        const int bcol = wc * 32 + j * 16 + ll;
        v8bf lo = *reinterpret_cast<const v8bf*>(&Blds[bcol * BS + 8 * lh]);
        v8bf hi = *reinterpret_cast<const v8bf*>(&Blds[bcol * BS + 16 + 8 * lh]);
        b[j] = __builtin_shufflevector(lo, hi, 0,1,2,3,4,5,6,7,8,9,10,11,12,13,14,15);
    }
    #pragma unroll
    for (int i = 0; i < 2; ++i)
        #pragma unroll
        for (int j = 0; j < 2; ++j)
            acc[i][j] = __builtin_amdgcn_wmma_f32_16x16x32_bf16(
                false, a[i], false, b[j], (short)0, acc[i][j], false, false);
}

// one pipeline stage: prefetch (kb_next -> nxt) around compute (cur)
static __device__ inline void gemm_step(const GemmCtx& c, int kb_next, bool more,
                                        bf16_t* cur, bf16_t* nxt,
                                        int wr, int wc, int ll, int lh,
                                        v8f acc[2][2]) {
    v4f va[4];
    if (more) issue_fill(c, kb_next, nxt, va);
    compute_step(cur, wr, wc, ll, lh, acc);
    if (more) store_fill(c, nxt, va);
    asm volatile("s_wait_asynccnt 0x0" ::: "memory");
    __syncthreads();
}

static __device__ inline void gemm_tile(const float* __restrict__ A, int lda,
                                        const bf16_t* __restrict__ Bm, int ldb,
                                        int K, int tileM, int tileN,
                                        bf16_t* lds, v8f acc[2][2]) {
    bf16_t* buf0 = lds;
    bf16_t* buf1 = lds + BUFE;

    const int tid  = threadIdx.x;
    const int lane = tid & 31;
    const int wave = tid >> 5;
    const int wr   = wave >> 1;
    const int wc   = wave & 1;
    const int ll   = lane & 15;
    const int lh   = lane >> 4;

    GemmCtx c;
    c.lda = (size_t)lda;
    c.ldb = (size_t)ldb;
    c.Abase = A + (size_t)(tileM + (tid >> 3)) * lda + (tid & 7) * 4;
    c.Bbase = Bm + (size_t)(tileN + (tid >> 2)) * ldb + (tid & 3) * 8;
    c.afill_off = (tid >> 3) * AS + (tid & 7) * 4;
    c.bfill_off = (tid >> 2) * BS + (tid & 3) * 8;

    // prologue: fill buffer 0 for kb=0
    {
        v4f va[4];
        issue_fill(c, 0, buf0, va);
        store_fill(c, buf0, va);
        asm volatile("s_wait_asynccnt 0x0" ::: "memory");
        __syncthreads();
    }
    // steady state: K/BK is even for every GEMM in this model
    for (int kb = 0; kb < K; kb += 2 * BK) {
        gemm_step(c, kb + BK, true,           buf0, buf1, wr, wc, ll, lh, acc);
        gemm_step(c, kb + 2 * BK, kb + 2 * BK < K, buf1, buf0, wr, wc, ll, lh, acc);
    }
}
// acc[i][j] element r @ lane: row = tileM + wr*32 + i*16 + (laneHi?8:0) + r
//                             col = tileN + wc*32 + j*16 + laneLo

// ------------------------------------------------------- transpose conv ---
// in: f32 [K][Nc] row-major  ->  out: bf16 [Nc][K] row-major. 32x32 tiles.
__global__ void __launch_bounds__(NTHREADS)
tconv_kernel(const float* __restrict__ in, bf16_t* __restrict__ out,
             int K, int Nc) {
    __shared__ alignas(16) bf16_t t[32 * 40];
    const int n0 = blockIdx.x * 32, k0 = blockIdx.y * 32;
    const int tid = threadIdx.x;
    {
        const int k = tid >> 3, n4 = tid & 7;
        v4f f = *(const v4f*)&in[(size_t)(k0 + k) * Nc + n0 + n4 * 4];
        v4bf bv = __builtin_convertvector(f, v4bf);
        #pragma unroll
        for (int j = 0; j < 4; ++j) t[(n4 * 4 + j) * 40 + k] = bv[j];
    }
    __syncthreads();
    {
        const int n = tid >> 3, k4 = tid & 7;
        *(v4bf*)&out[(size_t)(n0 + n) * K + k0 + k4 * 4] = *(v4bf*)&t[n * 40 + k4 * 4];
    }
}

// ------------------------------------------------------------ kernel 1 ----
__global__ void __launch_bounds__(NTHREADS)
qkv_gemm_kernel(const float* __restrict__ x, const bf16_t* __restrict__ wT,
                const float* __restrict__ bias, float* __restrict__ qws,
                bf16_t* __restrict__ kws, bf16_t* __restrict__ vws) {
    __shared__ alignas(16) bf16_t lds[2 * BUFE];
    const int tileN = blockIdx.x * BN;
    const int tileM = blockIdx.y * BM;
    v8f acc[2][2] = {};
    gemm_tile(x, Cc, wT, Cc, Cc, tileM, tileN, lds, acc);

    const int lane = threadIdx.x & 31, wave = threadIdx.x >> 5;
    const int wr = wave >> 1, wc = wave & 1, ll = lane & 15, lh = lane >> 4;
    #pragma unroll
    for (int j = 0; j < 2; ++j) {
        const int c = tileN + wc * 32 + j * 16 + ll;     // t uniform per subtile
        const int t = c / Cc, rem = c % Cc;
        const int h = rem / Dd, d = rem % Dd;
        const float bv = bias[c];
        #pragma unroll
        for (int i = 0; i < 2; ++i) {
            #pragma unroll
            for (int r = 0; r < 8; ++r) {
                const int m = tileM + wr * 32 + i * 16 + lh * 8 + r;
                const int b = m / Nn, n = m % Nn;
                const float val = acc[i][j][r] + bv;
                if (t == 0)
                    qws[((size_t)(b * Hh + h) * Nn + n) * Dd + d] = val;
                else if (t == 1)
                    kws[((size_t)(b * Hh + h) * Nn + n) * Dd + d] = (bf16_t)val;
                else
                    vws[((size_t)(b * Hh + h) * Dd + d) * Nn + n] = (bf16_t)val;
            }
        }
    }
}

// ------------------------------------------------------------ kernel 2 ----
__global__ void __launch_bounds__(NTHREADS)
scores_kernel(const float* __restrict__ q, const bf16_t* __restrict__ kten,
              const int* __restrict__ mask, const float* __restrict__ pbias,
              float* __restrict__ att) {
    __shared__ alignas(16) bf16_t lds[2 * BUFE];
    const int z = blockIdx.z;                 // b*H + h
    const int b = z / Hh, h = z % Hh;
    const int tileN = blockIdx.x * BN;
    const int tileM = blockIdx.y * BM;
    v8f acc[2][2] = {};
    gemm_tile(q + (size_t)z * Nn * Dd, Dd,
              kten + (size_t)z * Nn * Dd, Dd, Dd, tileM, tileN, lds, acc);

    const int lane = threadIdx.x & 31, wave = threadIdx.x >> 5;
    const int wr = wave >> 1, wc = wave & 1, ll = lane & 15, lh = lane >> 4;
    const float pb = pbias[h];
    #pragma unroll
    for (int j = 0; j < 2; ++j) {
        const int kk = tileN + wc * 32 + j * 16 + ll;
        const int mv = mask[b * Nn + kk];
        #pragma unroll
        for (int i = 0; i < 2; ++i) {
            #pragma unroll
            for (int r = 0; r < 8; ++r) {
                const int qq = tileM + wr * 32 + i * 16 + lh * 8 + r;
                float v = acc[i][j][r] * 0.125f + pb;
                if (mv == 0) v = -1e9f;
                att[((size_t)z * Nn + qq) * Nn + kk] = v;
            }
        }
    }
}

// ------------------------------------------------------------ kernel 3 ----
__global__ void __launch_bounds__(NTHREADS)
softmax_kernel(float* __restrict__ att) {
    __shared__ float red[NTHREADS];
    float* p = att + (size_t)blockIdx.x * Nn;
    const int tid = threadIdx.x;

    v4f va = *(const v4f*)&p[tid * 8];
    v4f vb = *(const v4f*)&p[tid * 8 + 4];
    float m = -3.4e38f;
    #pragma unroll
    for (int i = 0; i < 4; ++i) m = fmaxf(m, fmaxf(va[i], vb[i]));
    red[tid] = m; __syncthreads();
    for (int s = NTHREADS / 2; s > 0; s >>= 1) {
        if (tid < s) red[tid] = fmaxf(red[tid], red[tid + s]);
        __syncthreads();
    }
    m = red[0]; __syncthreads();

    float sum = 0.f;
    #pragma unroll
    for (int i = 0; i < 4; ++i) {
        va[i] = __expf(va[i] - m); sum += va[i];
        vb[i] = __expf(vb[i] - m); sum += vb[i];
    }
    red[tid] = sum; __syncthreads();
    for (int s = NTHREADS / 2; s > 0; s >>= 1) {
        if (tid < s) red[tid] += red[tid + s];
        __syncthreads();
    }
    const float inv = 1.f / red[0];
    #pragma unroll
    for (int i = 0; i < 4; ++i) { va[i] *= inv; vb[i] *= inv; }
    *(v4f*)&p[tid * 8] = va;
    *(v4f*)&p[tid * 8 + 4] = vb;
}

// ------------------------------------------------------------ kernel 4 ----
__global__ void __launch_bounds__(NTHREADS)
pv_kernel(const float* __restrict__ att, const bf16_t* __restrict__ vten,
          float* __restrict__ out_tmp) {
    __shared__ alignas(16) bf16_t lds[2 * BUFE];
    const int z = blockIdx.z;
    const int b = z / Hh, h = z % Hh;
    const int tileN = blockIdx.x * BN;                 // always 0 (Dd == BN)
    const int tileM = blockIdx.y * BM;
    v8f acc[2][2] = {};
    gemm_tile(att + (size_t)z * Nn * Nn, Nn,
              vten + (size_t)z * Dd * Nn, Nn, Nn, tileM, tileN, lds, acc);

    const int lane = threadIdx.x & 31, wave = threadIdx.x >> 5;
    const int wr = wave >> 1, wc = wave & 1, ll = lane & 15, lh = lane >> 4;
    #pragma unroll
    for (int j = 0; j < 2; ++j) {
        const int d = tileN + wc * 32 + j * 16 + ll;
        #pragma unroll
        for (int i = 0; i < 2; ++i) {
            #pragma unroll
            for (int r = 0; r < 8; ++r) {
                const int qq = tileM + wr * 32 + i * 16 + lh * 8 + r;
                out_tmp[(size_t)(b * Nn + qq) * Cc + h * Dd + d] = acc[i][j][r];
            }
        }
    }
}

// ------------------------------------------------------------ kernel 5 ----
__global__ void __launch_bounds__(NTHREADS)
proj_kernel(const float* __restrict__ inp, const bf16_t* __restrict__ wT,
            const float* __restrict__ bias, float* __restrict__ out) {
    __shared__ alignas(16) bf16_t lds[2 * BUFE];
    const int tileN = blockIdx.x * BN;
    const int tileM = blockIdx.y * BM;
    v8f acc[2][2] = {};
    gemm_tile(inp, Cc, wT, Cc, Cc, tileM, tileN, lds, acc);

    const int lane = threadIdx.x & 31, wave = threadIdx.x >> 5;
    const int wr = wave >> 1, wc = wave & 1, ll = lane & 15, lh = lane >> 4;
    #pragma unroll
    for (int j = 0; j < 2; ++j) {
        const int c = tileN + wc * 32 + j * 16 + ll;
        const float bv = bias[c];
        #pragma unroll
        for (int i = 0; i < 2; ++i) {
            #pragma unroll
            for (int r = 0; r < 8; ++r) {
                const int m = tileM + wr * 32 + i * 16 + lh * 8 + r;
                out[(size_t)m * Cc + c] = acc[i][j][r] + bv;
            }
        }
    }
}

// --------------------------------------------------------------- launch ---
extern "C" void kernel_launch(void* const* d_in, const int* in_sizes, int n_in,
                              void* d_out, int out_size, void* d_ws, size_t ws_size,
                              hipStream_t stream) {
    const float* x      = (const float*)d_in[0];
    const int*   mask   = (const int*)  d_in[1];
    const float* w_qkv  = (const float*)d_in[2];
    const float* b_qkv  = (const float*)d_in[3];
    const float* w_proj = (const float*)d_in[4];
    const float* b_proj = (const float*)d_in[5];
    const float* pbias  = (const float*)d_in[6];

    float* out = (float*)d_out;                       // [B,N,C]
    float* att = out + (size_t)Bb * Nn * Cc;          // [B,H,N,N]

    // workspace carve (~55 MB)
    float*  qws    = (float*)d_ws;                           // f32 [B,H,N,D]
    bf16_t* kws    = (bf16_t*)(qws + QKV_ELEMS);             // bf16 [B,H,N,D]
    bf16_t* vws    = kws + QKV_ELEMS;                        // bf16 [B,H,D,N]
    bf16_t* wqkvT  = vws + QKV_ELEMS;                        // bf16 [3C][C]
    bf16_t* wprojT = wqkvT + (size_t)3 * Cc * Cc;            // bf16 [C][C]
    float*  otmp   = qws;                                    // reuse q region

    const dim3 blk(NTHREADS);
    tconv_kernel<<<dim3(3 * Cc / 32, Cc / 32), blk, 0, stream>>>(w_qkv, wqkvT, Cc, 3 * Cc);
    tconv_kernel<<<dim3(Cc / 32, Cc / 32),     blk, 0, stream>>>(w_proj, wprojT, Cc, Cc);

    qkv_gemm_kernel<<<dim3(3 * Cc / BN, Bb * Nn / BM), blk, 0, stream>>>(
        x, wqkvT, b_qkv, qws, kws, vws);
    scores_kernel<<<dim3(Nn / BN, Nn / BM, Bb * Hh), blk, 0, stream>>>(
        qws, kws, mask, pbias, att);
    softmax_kernel<<<dim3(Bb * Hh * Nn), blk, 0, stream>>>(att);
    pv_kernel<<<dim3(Dd / BN, Nn / BM, Bb * Hh), blk, 0, stream>>>(att, vws, otmp);
    proj_kernel<<<dim3(Cc / BN, Bb * Nn / BM), blk, 0, stream>>>(otmp, wprojT, b_proj, out);
}
